// GATsep_18433999635062
// MI455X (gfx1250) — matrix operand order repaired and checked by reading, not compiled, not run on hardware
//
#include <hip/hip_runtime.h>
#include <cstdint>
#include <cstddef>

#define NEG_SLOPE 0.2f

typedef __attribute__((ext_vector_type(16))) _Float16 v16h;
typedef __attribute__((ext_vector_type(8)))  _Float16 v8h;
typedef __attribute__((ext_vector_type(8)))  float    v8f;

// ---------------------------------------------------------------------------
// f32 -> f16 row-major conversion (for GEMM A operands)
// ---------------------------------------------------------------------------
__global__ void f32_to_f16(const float* __restrict__ src,
                           _Float16* __restrict__ dst, long n)
{
  long i = (long)blockIdx.x * blockDim.x + threadIdx.x;
  if (i < n) dst[i] = (_Float16)src[i];
}

// ---------------------------------------------------------------------------
// Pack W[K,NOUT] f32 into B-fragment-friendly f16 layout:
//   chunk index t = (kt*NOUT + col)*2 + half  ->  16 f16, k = kt*32 + half*16 + j
// Each chunk is 32B contiguous => one v16h load per k-step in the GEMM.
// ---------------------------------------------------------------------------
__global__ void pack_w(const float* __restrict__ W, _Float16* __restrict__ Wp,
                       int K, int NOUT)
{
  int t = (int)(blockIdx.x * blockDim.x + threadIdx.x);
  int chunks = (K / 32) * NOUT * 2;
  if (t >= chunks) return;
  int half = t & 1;
  int col  = (t >> 1) % NOUT;
  int kt   = (t >> 1) / NOUT;
  _Float16* dst = Wp + (size_t)t * 16;
#pragma unroll
  for (int j = 0; j < 16; ++j)
    dst[j] = (_Float16)W[(size_t)(kt * 32 + half * 16 + j) * NOUT + col];
}

// ---------------------------------------------------------------------------
// GEMM: C[M,NOUT] = A16[M,K] @ W + bias, via v_wmma_f32_16x16x32_f16.
// One wave32 per 16x16 output tile. Fragment maps used by this kernel:
//   A 16x32 f16 : lane l -> row l&15 ; elem j -> k = (l>>4)*16 + j (contig)
//   B 32x16 f16 : lane l -> col l&15 ; elem j -> k = (l>>4)*16 + j (packed)
//   D 16x16 f32 : elem v -> row v + 8*(l>>4), col l&15
// ---------------------------------------------------------------------------
template <int K, int NOUT>
__global__ __launch_bounds__(256)
void gemm_wmma_f16(const _Float16* __restrict__ A16, int M,
                   const _Float16* __restrict__ Wp,
                   const float* __restrict__ bias,
                   float* __restrict__ C)
{
  int wave = (int)((blockIdx.x * blockDim.x + threadIdx.x) >> 5);
  int lane = (int)(threadIdx.x & 31);
  const int ntiles = NOUT >> 4;
  const int mtiles = (M + 15) >> 4;
  const int total  = mtiles * ntiles;
  if (wave >= total) wave = total - 1;     // clamp: EXEC stays all-ones (WMMA req)
  int tm   = wave / ntiles;
  int tn   = wave - tm * ntiles;
  int half = lane >> 4;
  int lm   = lane & 15;
  int row  = tm * 16 + lm;
  if (row >= M) row = M - 1;               // address clamp, no divergence
  int col  = tn * 16 + lm;

  const _Float16* arow = A16 + (size_t)row * K + half * 16;

  v8f acc = {};
#pragma unroll
  for (int kt = 0; kt < K / 32; ++kt) {
    v8h a0 = *(const v8h*)(arow + kt * 32);        // k = kt*32 + half*16 + 0..7
    v8h a1 = *(const v8h*)(arow + kt * 32 + 8);    // k = kt*32 + half*16 + 8..15
    v16h a = __builtin_shufflevector(a0, a1, 0, 1, 2, 3, 4, 5, 6, 7,
                                     8, 9, 10, 11, 12, 13, 14, 15);
    v16h b = *(const v16h*)(Wp + ((size_t)(kt * NOUT + col) * 2 + half) * 16);
    acc = __builtin_amdgcn_wmma_f32_16x16x32_f16(false, a, false, b,
                                                 (short)0, acc, false, false);
  }

  float bc = bias[col];
  float* cp = C + (size_t)(tm * 16 + half * 8) * NOUT + col;
  if (tm * 16 + 16 <= M) {
    // Wave-uniform full tile: unconditional coalesced stores (always taken
    // for M % 16 == 0) -> single scalar branch, no per-element exec masking.
#pragma unroll
    for (int v = 0; v < 8; ++v) cp[(size_t)v * NOUT] = acc[v] + bc;
  } else {
#pragma unroll
    for (int v = 0; v < 8; ++v) {
      int r = tm * 16 + v + half * 8;
      if (r < M) C[(size_t)r * NOUT + col] = acc[v] + bc;
    }
  }
}

// ---------------------------------------------------------------------------
// Utility fill
// ---------------------------------------------------------------------------
__global__ void fill_f32(float* __restrict__ p, float v, long n) {
  long i = (long)blockIdx.x * blockDim.x + threadIdx.x;
  if (i < n) p[i] = v;
}

// Ordered-float atomic max (init value must be -inf)
__device__ inline void atomicMaxF(float* addr, float val) {
  if (!(val < 0.0f)) atomicMax((int*)addr, __float_as_int(val));
  else               atomicMin((unsigned int*)addr, __float_as_uint(val));
}

// ---------------------------------------------------------------------------
// Edge pass 1: logits -> per-(tgt,head) running max.
// Thread per (edge, head). Edges e>=E are self loops (src=tgt=e-E).
// ---------------------------------------------------------------------------
template <int C>
__global__ void attn_logit_max(const float* __restrict__ xl,
                               const float* __restrict__ xr,
                               const int* __restrict__ src,
                               const int* __restrict__ tgt,
                               int E, int ET,
                               const float* __restrict__ att,
                               float* __restrict__ maxlog)
{
  long t = (long)blockIdx.x * blockDim.x + threadIdx.x;
  if (t >= (long)ET * 8) return;
  int e = (int)(t >> 3), h = (int)(t & 7);
  int s, g;
  if (e < E) { s = src[e]; g = tgt[e]; } else { s = e - E; g = s; }
  const float* pl = xl + (size_t)s * (8 * C) + h * C;
  const float* pr = xr + (size_t)g * (8 * C) + h * C;
  const float* pa = att + h * C;
  float acc = 0.0f;
#pragma unroll
  for (int c = 0; c < C; ++c) {
    float v = pl[c] + pr[c];
    v = (v > 0.0f) ? v : NEG_SLOPE * v;
    acc += v * pa[c];
  }
  atomicMaxF(&maxlog[(size_t)g * 8 + h], acc);
}

// ---------------------------------------------------------------------------
// Edge pass 2: p = exp(logit - max);  denom += p;  accum += p * xl[src]
// (softmax normalization folded into final division: sum(p*xl)/sum(p))
// ---------------------------------------------------------------------------
template <int C>
__global__ void attn_accum(const float* __restrict__ xl,
                           const float* __restrict__ xr,
                           const int* __restrict__ src,
                           const int* __restrict__ tgt,
                           int E, int ET,
                           const float* __restrict__ att,
                           const float* __restrict__ maxlog,
                           float* __restrict__ denom,
                           float* __restrict__ accum)
{
  long t = (long)blockIdx.x * blockDim.x + threadIdx.x;
  if (t >= (long)ET * 8) return;
  int e = (int)(t >> 3), h = (int)(t & 7);
  int s, g;
  if (e < E) { s = src[e]; g = tgt[e]; } else { s = e - E; g = s; }
  const float* pl = xl + (size_t)s * (8 * C) + h * C;
  const float* pr = xr + (size_t)g * (8 * C) + h * C;
  const float* pa = att + h * C;
  float acc = 0.0f;
#pragma unroll
  for (int c = 0; c < C; ++c) {
    float v = pl[c] + pr[c];
    v = (v > 0.0f) ? v : NEG_SLOPE * v;
    acc += v * pa[c];
  }
  float p = __expf(acc - maxlog[(size_t)g * 8 + h]);
  atomicAdd(&denom[(size_t)g * 8 + h], p);
  float* pd = accum + (size_t)g * (8 * C) + h * C;
#pragma unroll
  for (int c = 0; c < C; ++c) atomicAdd(&pd[c], p * pl[c]);
}

// ---------------------------------------------------------------------------
// Layer-1 finalize: mean over heads of accum/denom, +bias, ELU,
// write f16 into h2in16[:, 128:160].  Thread per (node, channel).
// ---------------------------------------------------------------------------
__global__ void finalize1(const float* __restrict__ accum,
                          const float* __restrict__ denom,
                          const float* __restrict__ bias,
                          _Float16* __restrict__ h2in16, int N)
{
  long t = (long)blockIdx.x * blockDim.x + threadIdx.x;
  if (t >= (long)N * 32) return;
  int i = (int)(t >> 5), c = (int)(t & 31);
  float s = 0.0f;
#pragma unroll
  for (int h = 0; h < 8; ++h)
    s += accum[(size_t)i * 256 + h * 32 + c] / denom[(size_t)i * 8 + h];
  s = s * 0.125f + bias[c];
  s = (s > 0.0f) ? s : (__expf(s) - 1.0f);     // ELU
  h2in16[(size_t)i * 160 + 128 + c] = (_Float16)s;
}

__global__ void copy_x16(const _Float16* __restrict__ x16,
                         _Float16* __restrict__ h2in16, int N)
{
  long t = (long)blockIdx.x * blockDim.x + threadIdx.x;
  if (t >= (long)N * 128) return;
  int i = (int)(t >> 7), c = (int)(t & 127);
  h2in16[(size_t)i * 160 + c] = x16[(size_t)i * 128 + c];
}

// ---------------------------------------------------------------------------
// Layer-2 finalize + log_softmax over 40 channels. Thread per node.
// ---------------------------------------------------------------------------
__global__ void finalize2(const float* __restrict__ accum,
                          const float* __restrict__ denom,
                          const float* __restrict__ bias,
                          float* __restrict__ out, int N)
{
  int i = (int)(blockIdx.x * blockDim.x + threadIdx.x);
  if (i >= N) return;
  float inv[8];
#pragma unroll
  for (int h = 0; h < 8; ++h) inv[h] = 1.0f / denom[(size_t)i * 8 + h];
  float v[40];
  float mx = -1e30f;
#pragma unroll
  for (int c = 0; c < 40; ++c) {
    float s = 0.0f;
#pragma unroll
    for (int h = 0; h < 8; ++h) s += accum[(size_t)i * 320 + h * 40 + c] * inv[h];
    s = s * 0.125f + bias[c];
    v[c] = s;
    mx = fmaxf(mx, s);
  }
  float se = 0.0f;
#pragma unroll
  for (int c = 0; c < 40; ++c) se += __expf(v[c] - mx);
  float lse = mx + __logf(se);
#pragma unroll
  for (int c = 0; c < 40; ++c) out[(size_t)i * 40 + c] = v[c] - lse;
}

// ---------------------------------------------------------------------------
// Host-side orchestration
// ---------------------------------------------------------------------------
static inline int cdiv(long a, long b) { return (int)((a + b - 1) / b); }

extern "C" void kernel_launch(void* const* d_in, const int* in_sizes, int n_in,
                              void* d_out, int out_size, void* d_ws, size_t ws_size,
                              hipStream_t stream)
{
  const float* x     = (const float*)d_in[0];
  const int*   ei    = (const int*)d_in[1];
  const float* Wl1   = (const float*)d_in[2];
  const float* bl1   = (const float*)d_in[3];
  const float* Wr1   = (const float*)d_in[4];
  const float* br1   = (const float*)d_in[5];
  const float* att1  = (const float*)d_in[6];
  const float* bias1 = (const float*)d_in[7];
  const float* Wl2   = (const float*)d_in[8];
  const float* bl2   = (const float*)d_in[9];
  const float* Wr2   = (const float*)d_in[10];
  const float* br2   = (const float*)d_in[11];
  const float* att2  = (const float*)d_in[12];
  const float* bias2 = (const float*)d_in[13];

  const int N  = in_sizes[0] / 128;   // 50000
  const int E  = in_sizes[1] / 2;     // 400000
  const int ET = E + N;               // edges + self loops
  const int* src = ei;
  const int* tgt = ei + E;

  // Workspace layout (256B-aligned regions, buffers reused across layers)
  char* ws = (char*)d_ws;
  const size_t szBig = (size_t)N * 320 * sizeof(float);            // 64 MB
  size_t off = 0;
  float*     xl      = (float*)(ws + off);  off += szBig;
  float*     xr      = (float*)(ws + off);  off += szBig;
  float*     accum   = (float*)(ws + off);  off += szBig;
  _Float16*  h2in16  = (_Float16*)(ws + off); off += (size_t)N * 160 * 2;
  _Float16*  x16     = (_Float16*)(ws + off); off += (size_t)N * 128 * 2;
  float*     maxlog  = (float*)(ws + off);  off += (size_t)N * 8 * 4;
  float*     denom   = (float*)(ws + off);  off += (size_t)N * 8 * 4;
  _Float16*  WpL     = (_Float16*)(ws + off); off += 160 * 320 * 2;
  _Float16*  WpR     = (_Float16*)(ws + off); off += 160 * 320 * 2;

  const int TB = 256;
  const long n8 = (long)N * 8;

  // Convert node features to f16 once (GEMM A operand for layer 1)
  f32_to_f16<<<cdiv((long)N * 128, TB), TB, 0, stream>>>(x, x16, (long)N * 128);

  // ================= Layer 1 =================
  {
    int chunks = (128 / 32) * 256 * 2;
    pack_w<<<cdiv(chunks, TB), TB, 0, stream>>>(Wl1, WpL, 128, 256);
    pack_w<<<cdiv(chunks, TB), TB, 0, stream>>>(Wr1, WpR, 128, 256);

    int tiles = ((N + 15) / 16) * (256 / 16);
    int blks  = cdiv(tiles, 8);
    gemm_wmma_f16<128, 256><<<blks, TB, 0, stream>>>(x16, N, WpL, bl1, xl);
    gemm_wmma_f16<128, 256><<<blks, TB, 0, stream>>>(x16, N, WpR, br1, xr);

    fill_f32<<<cdiv(n8, TB), TB, 0, stream>>>(maxlog, -__builtin_huge_valf(), n8);
    fill_f32<<<cdiv(n8, TB), TB, 0, stream>>>(denom, 0.0f, n8);
    fill_f32<<<cdiv((long)N * 256, TB), TB, 0, stream>>>(accum, 0.0f, (long)N * 256);

    long eth = (long)ET * 8;
    attn_logit_max<32><<<cdiv(eth, TB), TB, 0, stream>>>(xl, xr, src, tgt, E, ET,
                                                         att1, maxlog);
    attn_accum<32><<<cdiv(eth, TB), TB, 0, stream>>>(xl, xr, src, tgt, E, ET,
                                                     att1, maxlog, denom, accum);
    finalize1<<<cdiv((long)N * 32, TB), TB, 0, stream>>>(accum, denom, bias1,
                                                         h2in16, N);
    copy_x16<<<cdiv((long)N * 128, TB), TB, 0, stream>>>(x16, h2in16, N);
  }

  // ================= Layer 2 =================
  {
    int chunks = (160 / 32) * 320 * 2;
    pack_w<<<cdiv(chunks, TB), TB, 0, stream>>>(Wl2, WpL, 160, 320);
    pack_w<<<cdiv(chunks, TB), TB, 0, stream>>>(Wr2, WpR, 160, 320);

    int tiles = ((N + 15) / 16) * (320 / 16);
    int blks  = cdiv(tiles, 8);
    gemm_wmma_f16<160, 320><<<blks, TB, 0, stream>>>(h2in16, N, WpL, bl2, xl);
    gemm_wmma_f16<160, 320><<<blks, TB, 0, stream>>>(h2in16, N, WpR, br2, xr);

    fill_f32<<<cdiv(n8, TB), TB, 0, stream>>>(maxlog, -__builtin_huge_valf(), n8);
    fill_f32<<<cdiv(n8, TB), TB, 0, stream>>>(denom, 0.0f, n8);
    fill_f32<<<cdiv((long)N * 320, TB), TB, 0, stream>>>(accum, 0.0f, (long)N * 320);

    long eth = (long)ET * 8;
    attn_logit_max<40><<<cdiv(eth, TB), TB, 0, stream>>>(xl, xr, src, tgt, E, ET,
                                                         att2, maxlog);
    attn_accum<40><<<cdiv(eth, TB), TB, 0, stream>>>(xl, xr, src, tgt, E, ET,
                                                     att2, maxlog, denom, accum);
    finalize2<<<cdiv(N, TB), TB, 0, stream>>>(accum, denom, bias2, (float*)d_out, N);
  }
}
